// WaveFunction_47321949667616
// MI455X (gfx1250) — matrix operand order repaired and checked by reading, not compiled, not run on hardware
//
#include <hip/hip_runtime.h>
#include <hip/hip_bf16.h>
#include <math.h>

// ---------------------------------------------------------------------------
// Autoregressive wavefunction on MI455X (gfx1250, wave32).
// Per-site MLPs [2i,128,128,4] x 2 nets, batch 4096 -> ~70 GFLOP, compute
// bound (params fit in 192MB L2). Uses v_wmma_f32_16x16x32_f16 for layers
// 1 and 2; layer 3 (N=4) + mask/logsumexp epilogue done in VALU.
// X tiles staged with GLOBAL_LOAD_ASYNC_TO_LDS_B128 when the builtin exists.
// ---------------------------------------------------------------------------

typedef __attribute__((ext_vector_type(16))) _Float16 v16h;
typedef __attribute__((ext_vector_type(8)))  float    v8f;
typedef __attribute__((ext_vector_type(4)))  _Float16 v4h;

// Exact parameter type reported by clang for the async-LDS builtin:
// "__attribute__((__vector_size__(4 * sizeof(int)))) int __device__ *"
typedef int wf_v4i __attribute__((vector_size(16)));
typedef __attribute__((address_space(1))) wf_v4i* wf_gptr;   // global (AS1)
typedef __attribute__((address_space(3))) wf_v4i* wf_lptr;   // LDS (AS3)

#define SITES     128
#define HID       128
#define BATCH     4096
#define SPIN_UP   32
#define SPIN_DOWN 32
#define BT        128   // batch rows per block
#define SA        48    // LDS stride (halves) for X tile (96B rows, 16B aligned)
#define SB        40    // LDS stride (halves) for transposed W tile (80B rows)
#define SH        136   // LDS stride (halves) for H1 buffer (272B rows)

#if defined(__has_builtin)
# if __has_builtin(__builtin_amdgcn_global_load_async_to_lds_b128)
#  define WF_ASYNC_LDS 1
# endif
#endif
#ifndef WF_ASYNC_LDS
# define WF_ASYNC_LDS 0
#endif

// flat-params offset of site i (pytree order: per site (W1,b1),(W2,b2),(W3,b3))
__device__ __forceinline__ int site_base(int i) {
    // sum_{j<i} (2j*128 + 128 + 128*128 + 128 + 128*4 + 4) = 128*i*(i-1) + 17156*i
    return 128 * i * (i - 1) + 17156 * i;
}

__device__ __forceinline__ float silu(float v) {
    return v / (1.0f + __expf(-v));
}

__device__ __forceinline__ v8f wmma_f16(v16h a, v16h b, v8f c) {
    return __builtin_amdgcn_wmma_f32_16x16x32_f16(
        /*neg_a=*/false, a, /*neg_b=*/false, b,
        /*c_mod=*/(short)0, c, /*reuse_a=*/false, /*reuse_b=*/false);
}

// 16-byte global -> LDS copy; async direct-to-LDS when available.
__device__ __forceinline__ void copy16_g2l(const void* gsrc, void* ldst) {
#if WF_ASYNC_LDS
    __builtin_amdgcn_global_load_async_to_lds_b128(
        (wf_gptr)(uintptr_t)gsrc,   // AS1 global src (same numeric value)
        (wf_lptr)ldst,              // addrspacecast generic -> LDS
        /*offset=*/0, /*cpol=*/0);
#else
    uint4 q = *(const uint4*)gsrc;
    uint2* d = (uint2*)ldst;
    d[0] = make_uint2(q.x, q.y);
    d[1] = make_uint2(q.z, q.w);
#endif
}

__device__ __forceinline__ void async_fence() {
#if WF_ASYNC_LDS
    asm volatile("s_wait_asynccnt 0x0" ::: "memory");
#endif
}

// A fragment (16x32 f16): lane&15 = M row, (lane>>4)*8 = K half-offset.
// VGPR pairs: t0..3 -> K=klo+0..7 ; t4..7 -> K=16+klo+0..7
__device__ __forceinline__ v16h load_afrag(const _Float16* base, int stride,
                                           int m0, int k0, int lane) {
    const _Float16* rowp = base + (m0 + (lane & 15)) * stride + k0 + ((lane >> 4) * 8);
    v16h a;
    v4h* ap = (v4h*)&a;
    ap[0] = *(const v4h*)(rowp + 0);
    ap[1] = *(const v4h*)(rowp + 4);
    ap[2] = *(const v4h*)(rowp + 16);
    ap[3] = *(const v4h*)(rowp + 20);
    return a;
}

// B fragment (32x16 f16) from LDS stored transposed [N][K]:
// lane&15 = N col, (lane>>4)*8 = K offset; t0..3 -> K=klo+0..7 ; t4..7 -> K=16+klo+0..7
__device__ __forceinline__ v16h load_bfrag(const _Float16* wtile, int stride,
                                           int n0, int lane) {
    const _Float16* rowp = wtile + (n0 + (lane & 15)) * stride + ((lane >> 4) * 8);
    v16h b;
    v4h* bp = (v4h*)&b;
    bp[0] = *(const v4h*)(rowp + 0);
    bp[1] = *(const v4h*)(rowp + 4);
    bp[2] = *(const v4h*)(rowp + 16);
    bp[3] = *(const v4h*)(rowp + 20);
    return b;
}

// Stage one 32xK-slice of W (f32 row-major [K][N]) into LDS transposed [N][K]
// as f16, zero-padding rows >= Kt. Branch-free: Kt is even and kk is even, so
// validity is uniform per K-pair; pack two halves per 32-bit LDS store.
__device__ __forceinline__ void load_wtile(const float* __restrict__ W, int Kt,
                                           int k0, int tid, _Float16* wt) {
#pragma unroll
    for (int j = 0; j < 8; ++j) {
        int p  = tid + j * 256;        // 0..2047 pair index
        int n  = p & 127;
        int kk = (p >> 7) * 2;         // 0,2,...,30
        int k  = k0 + kk;
        int ks = (k < Kt) ? k : 0;     // clamp to a valid row
        const float* wp = W + (size_t)ks * HID + n;
        float w0 = wp[0];
        float w1 = wp[HID];
        bool ok = (k < Kt);
        w0 = ok ? w0 : 0.0f;
        w1 = ok ? w1 : 0.0f;
        union { _Float16 h[2]; unsigned u; } pk;
        pk.h[0] = (_Float16)w0;
        pk.h[1] = (_Float16)w1;
        *(unsigned*)(wt + n * SB + kk) = pk.u;
    }
}

// ---------------------------------------------------------------------------
// Prep: f16 feature matrix xh[B][256], exclusive spin cumsums, zero totals.
// ---------------------------------------------------------------------------
__global__ __launch_bounds__(256) void wf_prep_kernel(
        const int* __restrict__ x, _Float16* __restrict__ xh,
        unsigned* __restrict__ pref, float* __restrict__ ta,
        float* __restrict__ tp) {
    int b = blockIdx.x * blockDim.x + threadIdx.x;
    if (b >= BATCH) return;
    int ue = 0, de = 0;
    const int* xb = x + (size_t)b * SITES * 2;
    _Float16* xrow = xh + (size_t)b * 256;
    unsigned* prow = pref + (size_t)b * SITES;
    for (int i = 0; i < SITES; ++i) {
        prow[i] = (unsigned)(ue | (de << 16));
        int xu = xb[2 * i + 0];
        int xd = xb[2 * i + 1];
        xrow[2 * i + 0] = (_Float16)(float)xu;
        xrow[2 * i + 1] = (_Float16)(float)xd;
        ue += xu;
        de += xd;
    }
    ta[b] = 0.0f;
    tp[b] = 0.0f;
}

// ---------------------------------------------------------------------------
// Main: one block = (batch tile, site, net). MLP + epilogue.
// ---------------------------------------------------------------------------
__global__ __launch_bounds__(256) void wf_site_mlp_kernel(
        const float* __restrict__ amp_params, const float* __restrict__ phase_params,
        const _Float16* __restrict__ xh, const unsigned* __restrict__ pref,
        const int* __restrict__ x, float* __restrict__ total_a,
        float* __restrict__ total_p) {
    __shared__ _Float16 sh_xs[BT * SA];      // X tile (128 x 32), 16B-aligned rows
    __shared__ _Float16 sh_wt[HID * SB];     // W tile transposed [N][K] (128 x 32)
    __shared__ _Float16 sh_hs[BT * SH];      // H1 activations (128 x 128)
    __shared__ float    sh_part[BT * 4];     // layer-3 partial sums

    const int bt   = blockIdx.x;
    const int site = blockIdx.y;
    const int net  = blockIdx.z;
    const int tid  = threadIdx.x;
    const int wave = tid >> 5;
    const int lane = tid & 31;
    const int wm   = wave >> 1;   // 0..3 -> M offset wm*32
    const int wn   = wave & 1;    // 0..1 -> N offset wn*64

    const float* P  = net ? phase_params : amp_params;
    const int pb    = site_base(site);
    const int K1    = 2 * site;
    const float* W1 = P + pb;
    const float* b1 = W1 + 256 * site;
    const float* W2 = b1 + HID;
    const float* b2 = W2 + HID * HID;
    const float* W3 = b2 + HID;
    const float* b3 = W3 + HID * 4;

    const v8f vzero = {0.f, 0.f, 0.f, 0.f, 0.f, 0.f, 0.f, 0.f};
    v8f acc[2][4];
#pragma unroll
    for (int fm = 0; fm < 2; ++fm)
#pragma unroll
        for (int fn = 0; fn < 4; ++fn) acc[fm][fn] = vzero;

    // ---------------- Layer 1: X(128 x 2i) @ W1(2i x 128) -----------------
    const int xrow = tid >> 1;             // 2 threads per tile row
    const int xcol = (tid & 1) * 16;       // halves
    for (int k0 = 0; k0 < K1; k0 += 32) {
        // X tile: 32 bytes per thread, async direct-to-LDS (2 x b128)
        {
            const _Float16* src = xh + (size_t)(bt * BT + xrow) * 256 + k0 + xcol;
            _Float16* dst = sh_xs + xrow * SA + xcol;
            copy16_g2l(src, dst);
            copy16_g2l(src + 8, dst + 8);
        }
        // W1 tile: f32 -> f16, transposed into [N][K], zero-pad K >= 2i
        load_wtile(W1, K1, k0, tid, sh_wt);
        async_fence();
        __syncthreads();
        v16h a0 = load_afrag(sh_xs, SA, wm * 32 + 0, 0, lane);
        v16h a1 = load_afrag(sh_xs, SA, wm * 32 + 16, 0, lane);
#pragma unroll
        for (int fn = 0; fn < 4; ++fn) {
            v16h bf = load_bfrag(sh_wt, SB, wn * 64 + fn * 16, lane);
            acc[0][fn] = wmma_f16(a0, bf, acc[0][fn]);
            acc[1][fn] = wmma_f16(a1, bf, acc[1][fn]);
        }
        __syncthreads();
    }

    // bias + SiLU -> H1 (f16 in LDS), reset accumulators for layer 2
    {
#pragma unroll
        for (int fn = 0; fn < 4; ++fn) {
            int n = wn * 64 + fn * 16 + (lane & 15);
            float bb = b1[n];
#pragma unroll
            for (int fm = 0; fm < 2; ++fm) {
                int mrow = wm * 32 + fm * 16 + (lane >> 4) * 8;
#pragma unroll
                for (int r = 0; r < 8; ++r) {
                    float v = acc[fm][fn][r] + bb;
                    sh_hs[(mrow + r) * SH + n] = (_Float16)silu(v);
                }
                acc[fm][fn] = vzero;
            }
        }
    }
    __syncthreads();

    // ---------------- Layer 2: H1(128x128) @ W2(128x128) ------------------
#pragma unroll
    for (int kt = 0; kt < 4; ++kt) {
        int k0 = kt * 32;
        load_wtile(W2, HID, k0, tid, sh_wt);
        __syncthreads();
        v16h a0 = load_afrag(sh_hs, SH, wm * 32 + 0, k0, lane);
        v16h a1 = load_afrag(sh_hs, SH, wm * 32 + 16, k0, lane);
#pragma unroll
        for (int fn = 0; fn < 4; ++fn) {
            v16h bf = load_bfrag(sh_wt, SB, wn * 64 + fn * 16, lane);
            acc[0][fn] = wmma_f16(a0, bf, acc[0][fn]);
            acc[1][fn] = wmma_f16(a1, bf, acc[1][fn]);
        }
        __syncthreads();
    }

    // ---------------- Layer 3 (N=4): VALU + LDS reduction ------------------
    for (int idx = tid; idx < BT * 4; idx += 256) sh_part[idx] = 0.0f;
    __syncthreads();
    {
        float w3r[4][4];
        float b2v[4];
#pragma unroll
        for (int fn = 0; fn < 4; ++fn) {
            int n = wn * 64 + fn * 16 + (lane & 15);
            b2v[fn] = b2[n];
#pragma unroll
            for (int j = 0; j < 4; ++j) w3r[fn][j] = W3[n * 4 + j];
        }
#pragma unroll
        for (int fm = 0; fm < 2; ++fm) {
            int mrow = wm * 32 + fm * 16 + (lane >> 4) * 8;
#pragma unroll
            for (int r = 0; r < 8; ++r) {
                float ps0 = 0.f, ps1 = 0.f, ps2 = 0.f, ps3 = 0.f;
#pragma unroll
                for (int fn = 0; fn < 4; ++fn) {
                    float v = silu(acc[fm][fn][r] + b2v[fn]);
                    ps0 += v * w3r[fn][0];
                    ps1 += v * w3r[fn][1];
                    ps2 += v * w3r[fn][2];
                    ps3 += v * w3r[fn][3];
                }
                atomicAdd(&sh_part[(mrow + r) * 4 + 0], ps0);
                atomicAdd(&sh_part[(mrow + r) * 4 + 1], ps1);
                atomicAdd(&sh_part[(mrow + r) * 4 + 2], ps2);
                atomicAdd(&sh_part[(mrow + r) * 4 + 3], ps3);
            }
        }
    }
    __syncthreads();

    // ---------------- Epilogue: mask + logsumexp + gather ------------------
    if (tid < BT) {
        int m = tid;
        int b = bt * BT + m;
        float o[4];
#pragma unroll
        for (int j = 0; j < 4; ++j) o[j] = sh_part[m * 4 + j] + b3[j];

        int xu = x[((size_t)b * SITES + site) * 2 + 0];
        int xd = x[((size_t)b * SITES + site) * 2 + 1];

        if (net == 0) {
            unsigned pp = pref[(size_t)b * SITES + site];
            int ue = (int)(pp & 0xffffu);
            int de = (int)(pp >> 16);
            bool up0 = (site - ue) < (SITES - SPIN_UP);
            bool up1 = ue < SPIN_UP;
            bool dn0 = (site - de) < (SITES - SPIN_DOWN);
            bool dn1 = de < SPIN_DOWN;
            bool msk[4] = {up0 && dn0, up0 && dn1, up1 && dn0, up1 && dn1};
            float a4[4];
            float amax = -INFINITY;
#pragma unroll
            for (int t = 0; t < 4; ++t) {
                a4[t] = msk[t] ? o[t] : -INFINITY;
                amax = fmaxf(amax, a4[t]);
            }
            float s = 0.0f;
#pragma unroll
            for (int t = 0; t < 4; ++t) s += __expf(2.0f * (a4[t] - amax));
            float lse = amax + 0.5f * __logf(s);
            atomicAdd(&total_a[b], a4[xu * 2 + xd] - lse);
        } else {
            atomicAdd(&total_p[b], o[xu * 2 + xd]);
        }
    }
}

// ---------------------------------------------------------------------------
// Finalize: complex64 exp(a + i p) -> interleaved (re, im) floats.
// ---------------------------------------------------------------------------
__global__ __launch_bounds__(256) void wf_finalize_kernel(
        const float* __restrict__ ta, const float* __restrict__ tp,
        float* __restrict__ out) {
    int b = blockIdx.x * blockDim.x + threadIdx.x;
    if (b >= BATCH) return;
    float r = __expf(ta[b]);
    float ph = tp[b];
    out[2 * b + 0] = r * __cosf(ph);
    out[2 * b + 1] = r * __sinf(ph);
}

extern "C" void kernel_launch(void* const* d_in, const int* in_sizes, int n_in,
                              void* d_out, int out_size, void* d_ws, size_t ws_size,
                              hipStream_t stream) {
    const int*   x     = (const int*)d_in[0];
    const float* amp   = (const float*)d_in[1];
    const float* phase = (const float*)d_in[2];

    // workspace layout
    char* w = (char*)d_ws;
    _Float16* xh  = (_Float16*)w;                                // 2 MB
    unsigned* pr  = (unsigned*)(w + (size_t)2 * 1024 * 1024);    // 2 MB
    float*    ta  = (float*)(w + (size_t)4 * 1024 * 1024);       // 16 KB
    float*    tp  = ta + BATCH;                                  // 16 KB

    wf_prep_kernel<<<BATCH / 256, 256, 0, stream>>>(x, xh, pr, ta, tp);

    dim3 grid(BATCH / BT, SITES, 2);
    wf_site_mlp_kernel<<<grid, 256, 0, stream>>>(amp, phase, xh, pr, x, ta, tp);

    wf_finalize_kernel<<<BATCH / 256, 256, 0, stream>>>(ta, tp, (float*)d_out);
}
